// SuperGlue_58067957842193
// MI455X (gfx1250) — matrix operand order repaired and checked by reading, not compiled, not run on hardware
//
#include <hip/hip_runtime.h>
#include <hip/hip_bf16.h>
#include <math.h>

#define NPTS   2048
#define BATCH  2
#define DCH    256
#define NHEADS 4
#define HD     64
#define NP1    2049

typedef __attribute__((ext_vector_type(16))) __bf16 v16bf;
typedef __attribute__((ext_vector_type(8)))  float  v8f;
typedef __attribute__((ext_vector_type(4)))  unsigned int u32x4;

union FragBF { v16bf bf; unsigned short h[16]; u32x4 x4[2]; };
union AccF  { v8f v; float f[8]; };

__device__ __forceinline__ unsigned short f2bf(float x){
  union { float f; unsigned u; } c; c.f = x;
  unsigned u = c.u;
  unsigned r = (u + 0x7fffu + ((u >> 16) & 1u)) >> 16;
  return (unsigned short)r;
}

__device__ __forceinline__ void frag_zero(AccF &a){
#pragma unroll
  for (int i = 0; i < 8; ++i) a.f[i] = 0.f;
}

// A-operand (16x32 MxK bf16). Per documented layout the fragment is two
// contiguous 16B chunks at element offsets k0+8h and k0+8h+16.
__device__ __forceinline__ void load_a_bf(FragBF &a, const unsigned short* __restrict__ row,
                                          int k0, int lane){
  const unsigned short* p = row + k0 + 8*((lane >> 4) & 1);
  a.x4[0] = *(const u32x4*)p;
  a.x4[1] = *(const u32x4*)(p + 16);
}
// B-operand (32x16 KxN bf16): two 16B chunks at element offsets k0+8h and k0+8h+16.
__device__ __forceinline__ void load_b_bf(FragBF &b, const unsigned short* __restrict__ col,
                                          int k0, int lane){
  const unsigned short* p = col + k0 + 8*((lane >> 4) & 1);
  b.x4[0] = *(const u32x4*)p;
  b.x4[1] = *(const u32x4*)(p + 16);
}

__device__ __forceinline__ v8f wmma_bf16(const FragBF &a, const FragBF &b, v8f c){
  return __builtin_amdgcn_wmma_f32_16x16x32_bf16(false, a.bf, false, b.bf, (short)0, c,
                                                 false, false);
}

// ---------------------------------------------------------------------------
// fp32 -> bf16 bulk conversion (weights; runs once per call, L2-resident after)
// ---------------------------------------------------------------------------
__global__ void cvt_bf16_kernel(const float* __restrict__ src,
                                unsigned short* __restrict__ dst, int n){
  int i = blockIdx.x * blockDim.x + threadIdx.x;
  if (i < n) dst[i] = f2bf(src[i]);
}

// ---------------------------------------------------------------------------
// Keypoint encoder: 3 -> 32 -> 64 -> 128 -> 256, BN+ReLU on first 3 layers.
// ---------------------------------------------------------------------------
__global__ __launch_bounds__(256)
void kenc_kernel(const float* __restrict__ kpts, const float* __restrict__ scores,
                 const float* __restrict__ desc_in,
                 const float* __restrict__ c0W, const float* __restrict__ c0b,
                 const float* __restrict__ c1W, const float* __restrict__ c1b,
                 const float* __restrict__ c2W, const float* __restrict__ c2b,
                 const float* __restrict__ c3W, const float* __restrict__ c3b,
                 const float* __restrict__ bn0be, const float* __restrict__ bn0g,
                 const float* __restrict__ bn0m,  const float* __restrict__ bn0v,
                 const float* __restrict__ bn1be, const float* __restrict__ bn1g,
                 const float* __restrict__ bn1m,  const float* __restrict__ bn1v,
                 const float* __restrict__ bn2be, const float* __restrict__ bn2g,
                 const float* __restrict__ bn2m,  const float* __restrict__ bn2v,
                 float* __restrict__ descF, unsigned short* __restrict__ descB)
{
  __shared__ float s0[3], s1[32], s2[64], s3[128];
  int b = blockIdx.x / NPTS, n = blockIdx.x % NPTS;
  int tid = threadIdx.x;
  if (tid == 0){
    s0[0] = (kpts[((size_t)b*NPTS + n)*2 + 0] - 768.f) * (1.f/1075.2f);
    s0[1] = (kpts[((size_t)b*NPTS + n)*2 + 1] - 512.f) * (1.f/1075.2f);
    s0[2] = scores[(size_t)b*NPTS + n];
  }
  __syncthreads();
  if (tid < 32){
    float a = c0b[tid];
    for (int i = 0; i < 3; ++i) a += c0W[tid*3 + i] * s0[i];
    float sc = bn0g[tid] * rsqrtf(bn0v[tid] + 1e-5f);
    a = (a - bn0m[tid]) * sc + bn0be[tid];
    s1[tid] = fmaxf(a, 0.f);
  }
  __syncthreads();
  if (tid < 64){
    float a = c1b[tid];
    for (int i = 0; i < 32; ++i) a += c1W[tid*32 + i] * s1[i];
    float sc = bn1g[tid] * rsqrtf(bn1v[tid] + 1e-5f);
    a = (a - bn1m[tid]) * sc + bn1be[tid];
    s2[tid] = fmaxf(a, 0.f);
  }
  __syncthreads();
  if (tid < 128){
    float a = c2b[tid];
    for (int i = 0; i < 64; ++i) a += c2W[tid*64 + i] * s2[i];
    float sc = bn2g[tid] * rsqrtf(bn2v[tid] + 1e-5f);
    a = (a - bn2m[tid]) * sc + bn2be[tid];
    s3[tid] = fmaxf(a, 0.f);
  }
  __syncthreads();
  {
    float a = c3b[tid];
    for (int i = 0; i < 128; ++i) a += c3W[tid*128 + i] * s3[i];
    float out = desc_in[(size_t)b*DCH*NPTS + (size_t)tid*NPTS + n] + a;
    size_t idx = (size_t)b*NPTS*DCH + (size_t)n*DCH + tid;
    descF[idx] = out;
    descB[idx] = f2bf(out);
  }
}

// ---------------------------------------------------------------------------
// 1x1-conv GEMM, 2x2 register blocking (32x32 per wave), bf16 WMMA.
// out[c][n] = sum_k W[c][k] * X[n][k] + bias[c]; optional concat / BN+ReLU;
// output modes: 0 bf16 [b][n][cout]; 1 Q/K [b][head][n][64];
//               2 V [b][head][64][n]; 3 fp32 residual accum + bf16 mirror.
// ---------------------------------------------------------------------------
__global__ __launch_bounds__(256)
void gemm_conv(const unsigned short* __restrict__ Wbf, const float* __restrict__ bias,
               const unsigned short* __restrict__ in0, const unsigned short* __restrict__ in1,
               int cin, int cout, int split, int st0, int st1,
               const float* __restrict__ bn_g, const float* __restrict__ bn_be,
               const float* __restrict__ bn_m, const float* __restrict__ bn_v,
               int mode, unsigned short* __restrict__ outB, int out_stride, int out_bstride,
               float* __restrict__ resid)
{
  int b    = blockIdx.y;
  int lane = threadIdx.x & 31;
  int wave = threadIdx.x >> 5;
  int tm2  = cout >> 5;
  int tile = blockIdx.x * 8 + wave;
  int tm = (tile % tm2) * 2;      // 16-row units
  int tn = (tile / tm2) * 2;      // 16-col units
  if (tn >= (NPTS >> 4)) return;
  int col = lane & 15;
  const unsigned short* w0 = Wbf + (size_t)(tm*16 + col) * cin;
  const unsigned short* w1 = w0 + (size_t)16 * cin;
  const unsigned short* xA0 = in0 + (size_t)b*NPTS*st0 + (size_t)(tn*16 + col)*st0;
  const unsigned short* xA1 = xA0 + (size_t)16*st0;
  const unsigned short* xB0 = in1 ? (in1 + (size_t)b*NPTS*st1 + (size_t)(tn*16 + col)*st1)
                                  : xA0;
  const unsigned short* xB1 = xB0 + (size_t)16*st1;

  AccF a00, a01, a10, a11;
  frag_zero(a00); frag_zero(a01); frag_zero(a10); frag_zero(a11);

  for (int k0 = 0; k0 < split; k0 += 32){
    FragBF wa0, wa1, xb0, xb1;
    load_a_bf(wa0, w0, k0, lane);
    load_a_bf(wa1, w1, k0, lane);
    load_b_bf(xb0, xA0, k0, lane);
    load_b_bf(xb1, xA1, k0, lane);
    a00.v = wmma_bf16(wa0, xb0, a00.v);
    a01.v = wmma_bf16(wa0, xb1, a01.v);
    a10.v = wmma_bf16(wa1, xb0, a10.v);
    a11.v = wmma_bf16(wa1, xb1, a11.v);
    __builtin_prefetch(w0 + k0 + 32, 0, 1);
    __builtin_prefetch(w1 + k0 + 32, 0, 1);
  }
  for (int k0 = split; k0 < cin; k0 += 32){
    int kk = k0 - split;
    FragBF wa0, wa1, xb0, xb1;
    load_a_bf(wa0, w0, k0, lane);
    load_a_bf(wa1, w1, k0, lane);
    load_b_bf(xb0, xB0, kk, lane);
    load_b_bf(xb1, xB1, kk, lane);
    a00.v = wmma_bf16(wa0, xb0, a00.v);
    a01.v = wmma_bf16(wa0, xb1, a01.v);
    a10.v = wmma_bf16(wa1, xb0, a10.v);
    a11.v = wmma_bf16(wa1, xb1, a11.v);
  }

  int h = (lane >> 4) & 1;
  unsigned short* ob = outB + (size_t)b * out_bstride;
  auto epi = [&](AccF &acc, int ti, int tj){
    int nn = (tn + tj)*16 + col;
#pragma unroll
    for (int r = 0; r < 8; ++r){
      int mm = (tm + ti)*16 + r + 8*h;
      float val = acc.f[r] + bias[mm];
      if (bn_g){
        float sc = bn_g[mm] * rsqrtf(bn_v[mm] + 1e-5f);
        val = (val - bn_m[mm]) * sc + bn_be[mm];
        val = fmaxf(val, 0.f);
      }
      if (mode == 0){
        ob[(size_t)nn * out_stride + mm] = f2bf(val);
      } else if (mode == 1){
        ob[(size_t)(mm & 3) * (NPTS*HD) + (size_t)nn * HD + (mm >> 2)] = f2bf(val);
      } else if (mode == 2){
        ob[(size_t)(mm & 3) * (HD*NPTS) + (size_t)(mm >> 2) * NPTS + nn] = f2bf(val);
      } else {
        size_t idx = (size_t)b * NPTS * DCH + (size_t)nn * DCH + mm;
        float nv = resid[idx] + val;
        resid[idx] = nv;
        ob[(size_t)nn * out_stride + mm] = f2bf(nv);
      }
    }
  };
  epi(a00, 0, 0); epi(a01, 0, 1); epi(a10, 1, 0); epi(a11, 1, 1);
}

// ---------------------------------------------------------------------------
// Flash-style multi-head attention: 16 queries/wave, 32-key tiles,
// online softmax; logits and P*V via bf16 WMMA; P transposed through LDS.
// ---------------------------------------------------------------------------
__global__ __launch_bounds__(256)
void attn_kernel(const unsigned short* __restrict__ qB,
                 const unsigned short* __restrict__ kB,
                 const unsigned short* __restrict__ vB,
                 unsigned short* __restrict__ msgB)
{
  __shared__ unsigned short ldsP[8][16*32];
  int lane = threadIdx.x & 31;
  int wave = threadIdx.x >> 5;
  int head = blockIdx.y;
  int b    = blockIdx.z;
  int col  = lane & 15;
  int h    = (lane >> 4) & 1;
  size_t bo = (size_t)b * NPTS * DCH;
  const unsigned short* qh = qB + bo + (size_t)head * NPTS * HD;
  const unsigned short* kh = kB + bo + (size_t)head * NPTS * HD;
  const unsigned short* vh = vB + bo + (size_t)head * HD * NPTS;
  int q0 = (blockIdx.x * 8 + wave) * 16;

  FragBF aq0, aq1;
  load_a_bf(aq0, qh + (size_t)(q0 + col) * HD, 0,  lane);
  load_a_bf(aq1, qh + (size_t)(q0 + col) * HD, 32, lane);

  float runM[8], runL[8];
  AccF accm[4];
#pragma unroll
  for (int r = 0; r < 8; ++r){ runM[r] = -3.0e38f; runL[r] = 0.f; }
#pragma unroll
  for (int i = 0; i < 4; ++i) frag_zero(accm[i]);

  for (int m0 = 0; m0 < NPTS; m0 += 32){
    AccF s0, s1; frag_zero(s0); frag_zero(s1);
    {
      FragBF bk;
      load_b_bf(bk, kh + (size_t)(m0 + col) * HD, 0,  lane);
      s0.v = wmma_bf16(aq0, bk, s0.v);
      load_b_bf(bk, kh + (size_t)(m0 + col) * HD, 32, lane);
      s0.v = wmma_bf16(aq1, bk, s0.v);
      load_b_bf(bk, kh + (size_t)(m0 + 16 + col) * HD, 0,  lane);
      s1.v = wmma_bf16(aq0, bk, s1.v);
      load_b_bf(bk, kh + (size_t)(m0 + 16 + col) * HD, 32, lane);
      s1.v = wmma_bf16(aq1, bk, s1.v);
    }
#pragma unroll
    for (int r = 0; r < 8; ++r){
      float v0 = s0.f[r] * 0.125f;           // 1/sqrt(HD)
      float v1 = s1.f[r] * 0.125f;
      float tmax = fmaxf(v0, v1);
#pragma unroll
      for (int d = 1; d < 16; d <<= 1) tmax = fmaxf(tmax, __shfl_xor(tmax, d));
      float newM = fmaxf(runM[r], tmax);
      float sc = __expf(runM[r] - newM);
      float p0 = __expf(v0 - newM);
      float p1 = __expf(v1 - newM);
      float rs = p0 + p1;
#pragma unroll
      for (int d = 1; d < 16; d <<= 1) rs += __shfl_xor(rs, d);
      runL[r] = runL[r] * sc + rs;
      runM[r] = newM;
#pragma unroll
      for (int i = 0; i < 4; ++i) accm[i].f[r] *= sc;
      int qr = r + 8*h;
      ldsP[wave][qr*32 + col]       = f2bf(p0);
      ldsP[wave][qr*32 + 16 + col]  = f2bf(p1);
    }
    __syncthreads();
    FragBF ap;
    load_a_bf(ap, &ldsP[wave][col*32], 0, lane);
#pragma unroll
    for (int i = 0; i < 4; ++i){
      FragBF bv;
      load_b_bf(bv, vh + (size_t)(i*16 + col) * NPTS, m0, lane);
      accm[i].v = wmma_bf16(ap, bv, accm[i].v);
    }
    __syncthreads();
  }
  unsigned short* mb = msgB + bo;
#pragma unroll
  for (int i = 0; i < 4; ++i){
#pragma unroll
    for (int r = 0; r < 8; ++r){
      float val = accm[i].f[r] / runL[r];
      int n  = q0 + r + 8*h;
      int dd = i*16 + col;
      mb[(size_t)n * DCH + dd*4 + head] = f2bf(val);   // c = dd*HEADS + head
    }
  }
}

// ---------------------------------------------------------------------------
// Score GEMM (2x2 blocked): Z[b][n][m] = mdesc0[n]·mdesc1[m] / 16
// ---------------------------------------------------------------------------
__global__ __launch_bounds__(256)
void score_kernel(const unsigned short* __restrict__ md0,
                  const unsigned short* __restrict__ md1,
                  float* __restrict__ Z)
{
  int b = blockIdx.y;
  int lane = threadIdx.x & 31, wave = threadIdx.x >> 5;
  int tile = blockIdx.x * 8 + wave;
  int tm = (tile & 63) * 2, tn = (tile >> 6) * 2;
  int col = lane & 15, h = (lane >> 4) & 1;
  const unsigned short* r0 = md0 + (size_t)b*NPTS*DCH + (size_t)(tm*16 + col)*DCH;
  const unsigned short* r1 = r0 + (size_t)16*DCH;
  const unsigned short* c0 = md1 + (size_t)b*NPTS*DCH + (size_t)(tn*16 + col)*DCH;
  const unsigned short* c1 = c0 + (size_t)16*DCH;
  AccF a00, a01, a10, a11;
  frag_zero(a00); frag_zero(a01); frag_zero(a10); frag_zero(a11);
  for (int k0 = 0; k0 < DCH; k0 += 32){
    FragBF fa0, fa1, fb0, fb1;
    load_a_bf(fa0, r0, k0, lane);
    load_a_bf(fa1, r1, k0, lane);
    load_b_bf(fb0, c0, k0, lane);
    load_b_bf(fb1, c1, k0, lane);
    a00.v = wmma_bf16(fa0, fb0, a00.v);
    a01.v = wmma_bf16(fa0, fb1, a01.v);
    a10.v = wmma_bf16(fa1, fb0, a10.v);
    a11.v = wmma_bf16(fa1, fb1, a11.v);
  }
  float* zb = Z + (size_t)b*NP1*NP1;
  auto epi = [&](AccF &acc, int ti, int tj){
#pragma unroll
    for (int r = 0; r < 8; ++r){
      int nrow = (tm + ti)*16 + r + 8*h;
      int mcol = (tn + tj)*16 + col;
      zb[(size_t)nrow*NP1 + mcol] = acc.f[r] * 0.0625f;
    }
  };
  epi(a00, 0, 0); epi(a01, 0, 1); epi(a10, 1, 0); epi(a11, 1, 1);
}

__global__ void binfill_kernel(float* __restrict__ Z, const float* __restrict__ alpha_p){
  int idx = blockIdx.x * blockDim.x + threadIdx.x;
  int b = idx / NP1, i = idx % NP1;
  if (b >= BATCH) return;
  float alpha = alpha_p[0];
  float* zb = Z + (size_t)b*NP1*NP1;
  zb[(size_t)i*NP1 + NPTS] = alpha;
  zb[(size_t)NPTS*NP1 + i] = alpha;
}

__global__ void zero_kernel(float* __restrict__ p, int n){
  int i = blockIdx.x * blockDim.x + threadIdx.x;
  if (i < n) p[i] = 0.f;
}

// Sinkhorn row update: u[b,i] = log_mu[i] - LSE_j(Z[b,i,j] + v[b,j])
__global__ __launch_bounds__(256)
void sink_row_kernel(const float* __restrict__ Z, const float* __restrict__ v,
                     float* __restrict__ u)
{
  __shared__ float rm[256], rs[256];
  int b = blockIdx.x / NP1, i = blockIdx.x % NP1;
  const float* zr = Z + ((size_t)b*NP1 + i)*NP1;
  const float* vb = v + b*NP1;
  int tid = threadIdx.x;
  float m = -3.0e38f, s = 0.f;
  for (int j = tid; j < NP1; j += 256){
    float t = zr[j] + vb[j];
    if (t > m){ s = s * __expf(m - t) + 1.f; m = t; }
    else s += __expf(t - m);
  }
  rm[tid] = m; rs[tid] = s; __syncthreads();
  for (int off = 128; off > 0; off >>= 1){
    if (tid < off){
      float m2 = rm[tid+off], s2 = rs[tid+off], m1 = rm[tid], s1 = rs[tid];
      float mn = fmaxf(m1, m2);
      rm[tid] = mn; rs[tid] = s1*__expf(m1-mn) + s2*__expf(m2-mn);
    }
    __syncthreads();
  }
  if (tid == 0){
    float lse = rm[0] + __logf(rs[0]);
    float lmu = (i < NPTS) ? -8.317766167f : -0.6931471806f;  // -log4096 ; -log2
    u[b*NP1 + i] = lmu - lse;
  }
}

// Sinkhorn col update: v[b,j] = log_nu[j] - LSE_i(Z[b,i,j] + u[b,i])
__global__ __launch_bounds__(256)
void sink_col_kernel(const float* __restrict__ Z, const float* __restrict__ u,
                     float* __restrict__ v)
{
  int b = blockIdx.y;
  int j = blockIdx.x * 256 + threadIdx.x;
  if (j >= NP1) return;
  const float* zb = Z + (size_t)b*NP1*NP1;
  const float* ub = u + b*NP1;
  float m = -3.0e38f, s = 0.f;
  for (int i = 0; i < NP1; ++i){
    float t = zb[(size_t)i*NP1 + j] + ub[i];
    if (t > m){ s = s * __expf(m - t) + 1.f; m = t; }
    else s += __expf(t - m);
  }
  float lse = m + __logf(s);
  float lnu = (j < NPTS) ? -8.317766167f : -0.6931471806f;
  v[b*NP1 + j] = lnu - lse;
}

__global__ void final_scores_kernel(const float* __restrict__ Z, const float* __restrict__ u,
                                    const float* __restrict__ v, float* __restrict__ out){
  int idx = blockIdx.x * blockDim.x + threadIdx.x;
  if (idx >= BATCH*NP1*NP1) return;
  int b = idx / (NP1*NP1);
  int rem = idx % (NP1*NP1);
  int i = rem / NP1, j = rem % NP1;
  out[idx] = Z[idx] + u[b*NP1 + i] + v[b*NP1 + j] + 8.317766167f;  // -norm
}

__global__ __launch_bounds__(256)
void rowmax_kernel(const float* __restrict__ S, float* __restrict__ max0, int* __restrict__ i0){
  __shared__ float rm[256]; __shared__ int ri[256];
  int b = blockIdx.x / NPTS, i = blockIdx.x % NPTS;
  const float* sr = S + ((size_t)b*NP1 + i)*NP1;
  int tid = threadIdx.x;
  float m = -3.0e38f; int mi = 0;
  for (int j = tid; j < NPTS; j += 256){
    float t = sr[j];
    if (t > m){ m = t; mi = j; }
  }
  rm[tid] = m; ri[tid] = mi; __syncthreads();
  for (int off = 128; off > 0; off >>= 1){
    if (tid < off){
      if (rm[tid+off] > rm[tid] || (rm[tid+off] == rm[tid] && ri[tid+off] < ri[tid])){
        rm[tid] = rm[tid+off]; ri[tid] = ri[tid+off];
      }
    }
    __syncthreads();
  }
  if (tid == 0){ max0[b*NPTS + i] = rm[0]; i0[b*NPTS + i] = ri[0]; }
}

__global__ void colmax_kernel(const float* __restrict__ S, float* __restrict__ max1,
                              int* __restrict__ i1){
  int b = blockIdx.y;
  int j = blockIdx.x * 256 + threadIdx.x;
  if (j >= NPTS) return;
  const float* sb = S + (size_t)b*NP1*NP1;
  float m = -3.0e38f; int mi = 0;
  for (int i = 0; i < NPTS; ++i){
    float t = sb[(size_t)i*NP1 + j];
    if (t > m){ m = t; mi = i; }
  }
  max1[b*NPTS + j] = m; i1[b*NPTS + j] = mi;
}

__global__ void match_kernel(const float* __restrict__ max0, const int* __restrict__ i0,
                             const int* __restrict__ i1,
                             int* __restrict__ matches0, int* __restrict__ matches1,
                             float* __restrict__ ms0o, float* __restrict__ ms1o){
  int idx = blockIdx.x * blockDim.x + threadIdx.x;
  if (idx >= BATCH*NPTS) return;
  int b = idx / NPTS, n = idx % NPTS;
  const float* M0 = max0 + b*NPTS;
  const int* I0 = i0 + b*NPTS;
  const int* I1 = i1 + b*NPTS;
  bool mutual0 = (I1[I0[n]] == n);
  float ms0 = mutual0 ? __expf(M0[n]) : 0.f;
  bool valid0 = mutual0 && (ms0 > 0.2f);
  matches0[idx] = valid0 ? I0[n] : -1;
  ms0o[idx] = ms0;
  int jm = I1[n];
  bool mutual1 = (I0[jm] == n);
  float ms1 = 0.f;
  bool v0j = false;
  if (mutual1){
    bool mut0j = (I1[I0[jm]] == jm);
    float s = mut0j ? __expf(M0[jm]) : 0.f;
    ms1 = s;            // take_along_axis(ms0, i1)
    v0j = mut0j && (s > 0.2f);
  }
  matches1[idx] = (mutual1 && v0j) ? jm : -1;
  ms1o[idx] = mutual1 ? ms1 : 0.f;
}

// ---------------------------------------------------------------------------
extern "C" void kernel_launch(void* const* d_in, const int* in_sizes, int n_in,
                              void* d_out, int out_size, void* d_ws, size_t ws_size,
                              hipStream_t stream)
{
  (void)in_sizes; (void)n_in; (void)out_size; (void)ws_size;
  const float* kp0 = (const float*)d_in[0];
  const float* kp1 = (const float*)d_in[1];
  const float* sc0 = (const float*)d_in[2];
  const float* sc1 = (const float*)d_in[3];
  const float* de0 = (const float*)d_in[4];
  const float* de1 = (const float*)d_in[5];
  const float* bin = (const float*)d_in[6];
  const float* finW = (const float*)d_in[7];
  const float* finB = (const float*)d_in[8];
  const int KB = 9 + 16*18;   // kenc leaf base (sorted-key flatten)

  const float* bnp[3][4];
  for (int i = 0; i < 3; ++i)
    for (int j = 0; j < 4; ++j) bnp[i][j] = (const float*)d_in[KB + i*4 + j]; // be,g,m,v
  const float* cw[4]; const float* cb[4];
  for (int i = 0; i < 4; ++i){
    cw[i] = (const float*)d_in[KB + 12 + 2*i];
    cb[i] = (const float*)d_in[KB + 12 + 2*i + 1];
  }

  // workspace carve (identical every call -> deterministic)
  char* base = (char*)d_ws;
  size_t off = 0;
  auto carve = [&](size_t bytes)->void*{
    void* p = base + off; off = (off + bytes + 255) & ~(size_t)255; return p;
  };
  float* descF[2]; unsigned short *descB[2], *qB[2], *kBuf[2], *vBuf[2];
  unsigned short *msgB[2], *msg2B[2], *yB[2], *mdB[2];
  for (int s = 0; s < 2; ++s) descF[s] = (float*)carve((size_t)BATCH*NPTS*DCH*4);
  for (int s = 0; s < 2; ++s) descB[s] = (unsigned short*)carve((size_t)BATCH*NPTS*DCH*2);
  for (int s = 0; s < 2; ++s) qB[s]    = (unsigned short*)carve((size_t)BATCH*NPTS*DCH*2);
  for (int s = 0; s < 2; ++s) kBuf[s]  = (unsigned short*)carve((size_t)BATCH*NPTS*DCH*2);
  for (int s = 0; s < 2; ++s) vBuf[s]  = (unsigned short*)carve((size_t)BATCH*NPTS*DCH*2);
  for (int s = 0; s < 2; ++s) msgB[s]  = (unsigned short*)carve((size_t)BATCH*NPTS*DCH*2);
  for (int s = 0; s < 2; ++s) msg2B[s] = (unsigned short*)carve((size_t)BATCH*NPTS*DCH*2);
  for (int s = 0; s < 2; ++s) yB[s]    = (unsigned short*)carve((size_t)BATCH*NPTS*512*2);
  for (int s = 0; s < 2; ++s) mdB[s]   = (unsigned short*)carve((size_t)BATCH*NPTS*DCH*2);
  float* Z  = (float*)carve((size_t)BATCH*NP1*NP1*4);
  float* uv = (float*)carve((size_t)2*BATCH*NP1*4);
  float* u = uv, *v = uv + BATCH*NP1;
  float* max0w = (float*)carve((size_t)BATCH*NPTS*4);
  int*   i0w   = (int*)carve((size_t)BATCH*NPTS*4);
  float* max1w = (float*)carve((size_t)BATCH*NPTS*4);
  int*   i1w   = (int*)carve((size_t)BATCH*NPTS*4);

  // pre-convert conv weights to bf16 (L2-resident for the whole pass)
  auto cvt = [&](const float* src, int n)->const unsigned short*{
    unsigned short* dst = (unsigned short*)carve((size_t)n * 2);
    cvt_bf16_kernel<<<(n + 255)/256, 256, 0, stream>>>(src, dst, n);
    return dst;
  };

  // keypoint encoder + descriptor residual
  kenc_kernel<<<BATCH*NPTS, 256, 0, stream>>>(kp0, sc0, de0,
      cw[0],cb[0],cw[1],cb[1],cw[2],cb[2],cw[3],cb[3],
      bnp[0][0],bnp[0][1],bnp[0][2],bnp[0][3],
      bnp[1][0],bnp[1][1],bnp[1][2],bnp[1][3],
      bnp[2][0],bnp[2][1],bnp[2][2],bnp[2][3],
      descF[0], descB[0]);
  kenc_kernel<<<BATCH*NPTS, 256, 0, stream>>>(kp1, sc1, de1,
      cw[0],cb[0],cw[1],cb[1],cw[2],cb[2],cw[3],cb[3],
      bnp[0][0],bnp[0][1],bnp[0][2],bnp[0][3],
      bnp[1][0],bnp[1][1],bnp[1][2],bnp[1][3],
      bnp[2][0],bnp[2][1],bnp[2][2],bnp[2][3],
      descF[1], descB[1]);

  auto gemm = [&](const unsigned short* Wbf, const float* bias,
                  const unsigned short* in0, const unsigned short* in1,
                  int cin, int cout, int split, int st0, int st1,
                  const float* g, const float* be, const float* m_, const float* v_,
                  int mode, unsigned short* outB, int ostride, int obstride, float* resid){
    dim3 grid((cout >> 5) * (NPTS >> 5) / 8, BATCH);
    gemm_conv<<<grid, 256, 0, stream>>>(Wbf, bias, in0, in1, cin, cout, split, st0, st1,
                                        g, be, m_, v_, mode, outB, ostride, obstride, resid);
  };

  for (int l = 0; l < 18; ++l){
    int gb = 9 + 16*l;
    const float* mergeW = (const float*)d_in[gb+0];
    const float* mergeB = (const float*)d_in[gb+1];
    const float* mlp0W  = (const float*)d_in[gb+2];
    const float* mlp0B  = (const float*)d_in[gb+3];
    const float* mlp1W  = (const float*)d_in[gb+4];
    const float* mlp1B  = (const float*)d_in[gb+5];
    const float* bnbe   = (const float*)d_in[gb+6];
    const float* bng    = (const float*)d_in[gb+7];
    const float* bnm    = (const float*)d_in[gb+8];
    const float* bnv    = (const float*)d_in[gb+9];
    const float* p0W = (const float*)d_in[gb+10]; const float* p0B = (const float*)d_in[gb+11];
    const float* p1W = (const float*)d_in[gb+12]; const float* p1B = (const float*)d_in[gb+13];
    const float* p2W = (const float*)d_in[gb+14]; const float* p2B = (const float*)d_in[gb+15];

    const unsigned short* p0Wb = cvt(p0W, 256*256);
    const unsigned short* p1Wb = cvt(p1W, 256*256);
    const unsigned short* p2Wb = cvt(p2W, 256*256);
    const unsigned short* mgWb = cvt(mergeW, 256*256);
    const unsigned short* m0Wb = cvt(mlp0W, 512*512);
    const unsigned short* m1Wb = cvt(mlp1W, 256*512);

    bool cross = (l & 1);
    for (int s = 0; s < 2; ++s){
      const unsigned short* xq = descB[s];
      const unsigned short* xs = descB[cross ? 1 - s : s];
      gemm(p0Wb, p0B, xq, nullptr, 256, 256, 256, 256, 0,
           nullptr, nullptr, nullptr, nullptr, 1, qB[s],   0, NPTS*DCH, nullptr);
      gemm(p1Wb, p1B, xs, nullptr, 256, 256, 256, 256, 0,
           nullptr, nullptr, nullptr, nullptr, 1, kBuf[s], 0, NPTS*DCH, nullptr);
      gemm(p2Wb, p2B, xs, nullptr, 256, 256, 256, 256, 0,
           nullptr, nullptr, nullptr, nullptr, 2, vBuf[s], 0, NPTS*DCH, nullptr);
    }
    for (int s = 0; s < 2; ++s){
      dim3 ag(NPTS/128, NHEADS, BATCH);
      attn_kernel<<<ag, 256, 0, stream>>>(qB[s], kBuf[s], vBuf[s], msgB[s]);
    }
    for (int s = 0; s < 2; ++s){
      gemm(mgWb, mergeB, msgB[s], nullptr, 256, 256, 256, 256, 0,
           nullptr, nullptr, nullptr, nullptr, 0, msg2B[s], 256, NPTS*DCH, nullptr);
      gemm(m0Wb, mlp0B, descB[s], msg2B[s], 512, 512, 256, 256, 256,
           bng, bnbe, bnm, bnv, 0, yB[s], 512, NPTS*512, nullptr);
      gemm(m1Wb, mlp1B, yB[s], nullptr, 512, 256, 512, 512, 0,
           nullptr, nullptr, nullptr, nullptr, 3, descB[s], 256, NPTS*DCH, descF[s]);
    }
  }

  // final projection + score matrix
  const unsigned short* finWb = cvt(finW, 256*256);
  gemm(finWb, finB, descB[0], nullptr, 256, 256, 256, 256, 0,
       nullptr, nullptr, nullptr, nullptr, 0, mdB[0], 256, NPTS*DCH, nullptr);
  gemm(finWb, finB, descB[1], nullptr, 256, 256, 256, 256, 0,
       nullptr, nullptr, nullptr, nullptr, 0, mdB[1], 256, NPTS*DCH, nullptr);
  score_kernel<<<dim3(64*64/8, BATCH), 256, 0, stream>>>(mdB[0], mdB[1], Z);
  binfill_kernel<<<(BATCH*NP1 + 255)/256, 256, 0, stream>>>(Z, bin);

  // Sinkhorn (Z stays L2-resident: 33.6 MB << 192 MB)
  zero_kernel<<<(2*BATCH*NP1 + 255)/256, 256, 0, stream>>>(uv, 2*BATCH*NP1);
  for (int it = 0; it < 100; ++it){
    sink_row_kernel<<<BATCH*NP1, 256, 0, stream>>>(Z, v, u);
    sink_col_kernel<<<dim3((NP1 + 255)/256, BATCH), 256, 0, stream>>>(Z, u, v);
  }

  float* Sout = (float*)d_out + 4*BATCH*NPTS;
  final_scores_kernel<<<(BATCH*NP1*NP1 + 255)/256, 256, 0, stream>>>(Z, u, v, Sout);

  rowmax_kernel<<<BATCH*NPTS, 256, 0, stream>>>(Sout, max0w, i0w);
  colmax_kernel<<<dim3((NPTS + 255)/256, BATCH), 256, 0, stream>>>(Sout, max1w, i1w);

  int* m0o = (int*)d_out;
  int* m1o = m0o + BATCH*NPTS;
  float* ms0o = (float*)d_out + 2*BATCH*NPTS;
  float* ms1o = ms0o + BATCH*NPTS;
  match_kernel<<<(BATCH*NPTS + 255)/256, 256, 0, stream>>>(max0w, i0w, i1w,
                                                           m0o, m1o, ms0o, ms1o);
}
